// Seq2Seq_28192165330953
// MI455X (gfx1250) — compile-verified, matrix operand order checked by reading, not run on hardware
//
#include <hip/hip_runtime.h>
#include <hip/hip_bf16.h>
#include <stdint.h>

typedef _Float16 f16;
typedef __attribute__((ext_vector_type(16))) _Float16 v16h;
typedef __attribute__((ext_vector_type(8)))  _Float16 v8h;
typedef __attribute__((ext_vector_type(8)))  float    v8f;

#define WMMA_F32_F16(a, b, c) \
  __builtin_amdgcn_wmma_f32_16x16x32_f16(false, (a), false, (b), (short)0, (c), false, false)

static __device__ __forceinline__ float sigmoidf_(float x) {
  return 1.0f / (1.0f + __expf(-x));
}

// CDNA5 async DMA: copy 16 bytes global -> LDS without a VGPR round trip.
// Tracked with ASYNCcnt (ISA 10.x / 15.18.3 op 98). LDS operand = low 32 bits
// of the generic shared pointer (flat->LDS address truncation rule).
static __device__ __forceinline__ void async_copy16(const f16* gsrc, f16* ldst) {
  unsigned lds = (unsigned)(uintptr_t)ldst;
  asm volatile("global_load_async_to_lds_b128 %0, %1, off"
               :: "v"(lds), "v"(gsrc)
               : "memory");
}
static __device__ __forceinline__ void wait_async0() {
  asm volatile("s_wait_asynccnt 0x0" ::: "memory");
}

// Load a 16x32 f16 A/B fragment from a row-major matrix (leading dim `ld`,
// K contiguous). Row r0+lane%16; K mapping per CDNA5 ISA 7.12.2:
//   K(lane, e) = k0 + (e>=8 ? 16 : 0) + (lane>=16 ? 8 : 0) + (e & 7)
// -> two contiguous 16-byte loads per lane (global_load_b128 or ds_load_b128).
static __device__ __forceinline__ v16h load_frag(const f16* p, int ld, int r0, int k0) {
  const int lane = threadIdx.x & 31;
  const f16* rp = p + (size_t)(r0 + (lane & 15)) * (size_t)ld + k0 + ((lane >> 4) << 3);
  v8h lo = *(const v8h*)(rp);
  v8h hi = *(const v8h*)(rp + 16);
  v16h a;
#pragma unroll
  for (int i = 0; i < 8; ++i) { a[i] = lo[i]; a[i + 8] = hi[i]; }
  return a;
}

// ---------------------------------------------------------------------------
// fp32 -> f16 conversion, output zero-init
// ---------------------------------------------------------------------------
__global__ void f32_to_f16_kernel(const float* __restrict__ s, f16* __restrict__ d, int n) {
  int i = blockIdx.x * blockDim.x + threadIdx.x;
  if (i < n) d[i] = (f16)s[i];
}

__global__ void zero_kernel(float* __restrict__ p, int n) {
  int i = blockIdx.x * blockDim.x + threadIdx.x;
  if (i < n) p[i] = 0.f;
}

// ---------------------------------------------------------------------------
// Gate-fused LSTM layer step. Each wave owns (m0, j0) and computes the four
// gate tiles i/f/g/o with one shared A fragment per K-step (4x A reuse).
// A operands (x, h) come from LDS; h16 is double-buffered (read hcur, write
// hnext) so a single trailing barrier suffices.
// ---------------------------------------------------------------------------
static __device__ __forceinline__ void lstm_fused(
    const f16* xin, int Din, const f16* hcur, f16* hnext,
    const f16* __restrict__ Wih, const f16* __restrict__ Whh,
    const float* __restrict__ bih, const float* __restrict__ bhh,
    float* __restrict__ h, float* __restrict__ c) {
  const int wave = threadIdx.x >> 5;
  const int nwaves = blockDim.x >> 5;
  const int lane = threadIdx.x & 31;
  for (int t = wave; t < 64; t += nwaves) {  // 2 x 32 fused tiles, uniform
    const int m0 = (t & 1) * 16;
    const int j0 = (t >> 1) * 16;
    v8f ai = {}, af = {}, ag = {}, ao = {};
#pragma unroll 2
    for (int k = 0; k < Din; k += 32) {
      v16h a = load_frag(xin, Din, m0, k);
      ai = WMMA_F32_F16(a, load_frag(Wih, Din, j0, k), ai);
      af = WMMA_F32_F16(a, load_frag(Wih, Din, 512 + j0, k), af);
      ag = WMMA_F32_F16(a, load_frag(Wih, Din, 1024 + j0, k), ag);
      ao = WMMA_F32_F16(a, load_frag(Wih, Din, 1536 + j0, k), ao);
    }
#pragma unroll 2
    for (int k = 0; k < 512; k += 32) {
      v16h a = load_frag(hcur, 512, m0, k);
      ai = WMMA_F32_F16(a, load_frag(Whh, 512, j0, k), ai);
      af = WMMA_F32_F16(a, load_frag(Whh, 512, 512 + j0, k), af);
      ag = WMMA_F32_F16(a, load_frag(Whh, 512, 1024 + j0, k), ag);
      ao = WMMA_F32_F16(a, load_frag(Whh, 512, 1536 + j0, k), ao);
    }
    // lane holds gates i/f/g/o for the same (m, nn): full LSTM cell in regs.
    const int nn = j0 + (lane & 15);
    const int mb = m0 + ((lane >> 4) << 3);
    const float bi_ = bih[nn] + bhh[nn];
    const float bf_ = bih[512 + nn] + bhh[512 + nn];
    const float bg_ = bih[1024 + nn] + bhh[1024 + nn];
    const float bo_ = bih[1536 + nn] + bhh[1536 + nn];
#pragma unroll
    for (int r = 0; r < 8; ++r) {
      const int m = mb + r;
      float iv = sigmoidf_(ai[r] + bi_);
      float fv = sigmoidf_(af[r] + bf_);
      float gv = tanhf(ag[r] + bg_);
      float ov = sigmoidf_(ao[r] + bo_);
      float cn = fv * c[m * 512 + nn] + iv * gv;
      float hn = ov * tanhf(cn);
      c[m * 512 + nn] = cn;
      h[m * 512 + nn] = hn;
      hnext[m * 512 + nn] = (f16)hn;  // double buffer: no WAR vs hcur readers
    }
  }
  __threadfence_block();
  __syncthreads();
}

__global__ void __launch_bounds__(512) encoder_kernel(
    const f16* __restrict__ emb16, const int* __restrict__ src,
    const int* __restrict__ trg, const f16* __restrict__ Wih0,
    const f16* __restrict__ Whh0, const f16* __restrict__ Wih1,
    const f16* __restrict__ Whh1, const float* __restrict__ bih0,
    const float* __restrict__ bhh0, const float* __restrict__ bih1,
    const float* __restrict__ bhh1, float* __restrict__ h_st,
    float* __restrict__ c_st, float* __restrict__ h_dec,
    f16* __restrict__ h16g0, f16* __restrict__ h16g1, int* __restrict__ tok) {
  extern __shared__ f16 smem[];  // x16 + 2x double-buffered h16 = 160 KB
  f16* x16s = smem;              // 32*512 (ld 256 used for layer 0 input)
  f16* h0a = smem + 16384;
  f16* h0b = h0a + 16384;
  f16* h1a = h0b + 16384;
  f16* h1b = h1a + 16384;
  const int tid = threadIdx.x;
  for (int i = tid; i < 2 * 32 * 512; i += blockDim.x) { h_st[i] = 0.f; c_st[i] = 0.f; }
  for (int i = tid; i < 16384; i += blockDim.x) { h0a[i] = (f16)0.f; h1a[i] = (f16)0.f; }
  if (tid < 32) tok[tid] = trg[tid];  // initial decoder token = trg[0]
  __threadfence_block();
  __syncthreads();
  for (int s = 0; s < 64; ++s) {
    f16* h0c = (s & 1) ? h0b : h0a;
    f16* h0n = (s & 1) ? h0a : h0b;
    f16* h1c = (s & 1) ? h1b : h1a;
    f16* h1n = (s & 1) ? h1a : h1b;
    // Embedding gather: async DMA rows of the f16 table straight into LDS.
    for (int i = tid; i < 1024; i += blockDim.x) {  // 32 rows x 32 16B chunks
      const int b = i >> 5, ch = i & 31;
      async_copy16(emb16 + (size_t)src[s * 32 + b] * 256 + ch * 8,
                   x16s + b * 256 + ch * 8);
    }
    wait_async0();
    __syncthreads();
    lstm_fused(x16s, 256, h0c, h0n, Wih0, Whh0, bih0, bhh0, h_st, c_st);
    lstm_fused(h0n, 512, h1c, h1n, Wih1, Whh1, bih1, bhh1, h_st + 32 * 512,
               c_st + 32 * 512);
  }
  // Hand off to decoder: fp32 hidden + f16 copies (async-staged next kernel).
  for (int i = tid; i < 32 * 512; i += blockDim.x) {
    h_dec[i] = h_st[i];
    h_dec[32 * 512 + i] = h_st[32 * 512 + i];
    h16g0[i] = (f16)h_st[i];
    h16g1[i] = (f16)h_st[32 * 512 + i];
  }
}

// ---------------------------------------------------------------------------
// Gate-fused GRU layer step: accumulators r, z (gi+gh fused), gi_n, gh_n ->
// n = tanh(gi_n + r * gh_n) entirely in registers. Previous-step h16 arrives
// in LDS via async DMA (staged at entry); this writes the new h16 to LDS + global.
// ---------------------------------------------------------------------------
static __device__ __forceinline__ void gru_fused(
    const f16* xin, int Din, f16* hcur, f16* hnext,
    const f16* __restrict__ Wih, const f16* __restrict__ Whh,
    const float* __restrict__ bih, const float* __restrict__ bhh,
    float* __restrict__ h, f16* __restrict__ h16g) {
  const int tid = threadIdx.x;
  const int wave = tid >> 5;
  const int nwaves = blockDim.x >> 5;
  const int lane = tid & 31;
  // Stage previous h16 (written to global by the prior step) into LDS: pure
  // copy -> CDNA5 async-to-LDS path, 32 KB in 16B per-lane chunks.
  for (int i = tid; i < 2048; i += blockDim.x)
    async_copy16(h16g + i * 8, hcur + i * 8);
  wait_async0();
  __syncthreads();
  for (int t = wave; t < 64; t += nwaves) {
    const int m0 = (t & 1) * 16;
    const int j0 = (t >> 1) * 16;
    v8f ar = {}, az = {}, agi = {}, agh = {};
#pragma unroll 2
    for (int k = 0; k < Din; k += 32) {
      v16h a = load_frag(xin, Din, m0, k);
      ar = WMMA_F32_F16(a, load_frag(Wih, Din, j0, k), ar);
      az = WMMA_F32_F16(a, load_frag(Wih, Din, 512 + j0, k), az);
      agi = WMMA_F32_F16(a, load_frag(Wih, Din, 1024 + j0, k), agi);
    }
#pragma unroll 2
    for (int k = 0; k < 512; k += 32) {
      v16h a = load_frag(hcur, 512, m0, k);
      ar = WMMA_F32_F16(a, load_frag(Whh, 512, j0, k), ar);
      az = WMMA_F32_F16(a, load_frag(Whh, 512, 512 + j0, k), az);
      agh = WMMA_F32_F16(a, load_frag(Whh, 512, 1024 + j0, k), agh);
    }
    const int nn = j0 + (lane & 15);
    const int mb = m0 + ((lane >> 4) << 3);
    const float br_ = bih[nn] + bhh[nn];
    const float bz_ = bih[512 + nn] + bhh[512 + nn];
    const float bin_ = bih[1024 + nn];
    const float bhn_ = bhh[1024 + nn];
#pragma unroll
    for (int r = 0; r < 8; ++r) {
      const int m = mb + r;
      float rv = sigmoidf_(ar[r] + br_);
      float zv = sigmoidf_(az[r] + bz_);
      float nv = tanhf(agi[r] + bin_ + rv * (agh[r] + bhn_));
      float hn = (1.f - zv) * nv + zv * h[m * 512 + nn];
      h[m * 512 + nn] = hn;
      hnext[m * 512 + nn] = (f16)hn;  // LDS copy for next layer
      h16g[m * 512 + nn] = (f16)hn;   // global copy for fc + next step's DMA
    }
  }
  __threadfence_block();
  __syncthreads();
}

__global__ void __launch_bounds__(512) dec_gru_kernel(
    const f16* __restrict__ emb16, const int* __restrict__ tok,
    const f16* __restrict__ Wih0, const f16* __restrict__ Whh0,
    const f16* __restrict__ Wih1, const f16* __restrict__ Whh1,
    const float* __restrict__ bih0, const float* __restrict__ bhh0,
    const float* __restrict__ bih1, const float* __restrict__ bhh1,
    float* __restrict__ h_dec, f16* __restrict__ h16g0, f16* __restrict__ h16g1) {
  extern __shared__ f16 smem[];  // 160 KB
  f16* x16s = smem;              // 32*256 used
  f16* h0c = smem + 16384;
  f16* h0n = h0c + 16384;
  f16* h1c = h0n + 16384;
  f16* h1n = h1c + 16384;
  const int tid = threadIdx.x;
  // Token embedding gather via async DMA (f16 table -> LDS).
  for (int i = tid; i < 1024; i += blockDim.x) {  // 32 rows x 32 16B chunks
    const int b = i >> 5, ch = i & 31;
    async_copy16(emb16 + (size_t)tok[b] * 256 + ch * 8, x16s + b * 256 + ch * 8);
  }
  wait_async0();
  __syncthreads();
  gru_fused(x16s, 256, h0c, h0n, Wih0, Whh0, bih0, bhh0, h_dec, h16g0);
  gru_fused(h0n, 512, h1c, h1n, Wih1, Whh1, bih1, bhh1, h_dec + 32 * 512, h16g1);
}

// ---------------------------------------------------------------------------
// fc: [32x512] @ [512x32000]^T, grid-parallel. 4000 tiles, 1 per wave.
// fc_W16 (32 MB) is L2-resident on MI455X (192 MB L2).
// ---------------------------------------------------------------------------
__global__ void __launch_bounds__(256) fc_kernel(const f16* __restrict__ h16,
                                                 const f16* __restrict__ fcW16,
                                                 const float* __restrict__ fcb,
                                                 float* __restrict__ out) {
  const int t = blockIdx.x * (blockDim.x >> 5) + (threadIdx.x >> 5);
  if (t >= 4000) return;  // whole-wave uniform
  const int m0 = (t & 1) * 16;
  const int n0 = (t >> 1) * 16;
  v8f acc = {};
#pragma unroll 4
  for (int k = 0; k < 512; k += 32) {
    v16h a = load_frag(h16, 512, m0, k);
    v16h b = load_frag(fcW16, 512, n0, k);
    acc = WMMA_F32_F16(a, b, acc);
  }
  const int lane = threadIdx.x & 31;
  const int nn = n0 + (lane & 15);
  const int mb = m0 + ((lane >> 4) << 3);
  const float bias = fcb[nn];
#pragma unroll
  for (int r = 0; r < 8; ++r) out[(size_t)(mb + r) * 32000 + nn] = acc[r] + bias;
}

// ---------------------------------------------------------------------------
// argmax over V=32000 per batch row + teacher-forcing token select
// ---------------------------------------------------------------------------
__global__ void __launch_bounds__(256) argmax_kernel(const float* __restrict__ pred,
                                                     const int* __restrict__ trg,
                                                     const int* __restrict__ tf_mask,
                                                     int* __restrict__ tok, int step) {
  __shared__ float sv[256];
  __shared__ int si[256];
  const int b = blockIdx.x;
  const int tid = threadIdx.x;
  const float* row = pred + (size_t)b * 32000;
  float best = -3.4e38f;
  int bi = 0x7fffffff;
  for (int v = tid; v < 32000; v += 256) {
    float x = row[v];
    if (x > best || (x == best && v < bi)) { best = x; bi = v; }
  }
  sv[tid] = best;
  si[tid] = bi;
  __syncthreads();
  for (int s = 128; s > 0; s >>= 1) {
    if (tid < s) {
      if (sv[tid + s] > sv[tid] || (sv[tid + s] == sv[tid] && si[tid + s] < si[tid])) {
        sv[tid] = sv[tid + s];
        si[tid] = si[tid + s];
      }
    }
    __syncthreads();
  }
  if (tid == 0) tok[b] = (tf_mask[step] > 0) ? trg[step * 32 + b] : si[0];
}

// ---------------------------------------------------------------------------
extern "C" void kernel_launch(void* const* d_in, const int* in_sizes, int n_in,
                              void* d_out, int out_size, void* d_ws, size_t ws_size,
                              hipStream_t stream) {
  (void)in_sizes; (void)n_in; (void)out_size; (void)ws_size;
  // Inputs in setup_inputs() dict order
  const float* enc_emb = (const float*)d_in[0];
  const float* eWih0 = (const float*)d_in[1];
  const float* eWhh0 = (const float*)d_in[2];
  const float* ebih0 = (const float*)d_in[3];
  const float* ebhh0 = (const float*)d_in[4];
  const float* eWih1 = (const float*)d_in[5];
  const float* eWhh1 = (const float*)d_in[6];
  const float* ebih1 = (const float*)d_in[7];
  const float* ebhh1 = (const float*)d_in[8];
  const float* dec_emb = (const float*)d_in[9];
  const float* dWih0 = (const float*)d_in[10];
  const float* dWhh0 = (const float*)d_in[11];
  const float* dbih0 = (const float*)d_in[12];
  const float* dbhh0 = (const float*)d_in[13];
  const float* dWih1 = (const float*)d_in[14];
  const float* dWhh1 = (const float*)d_in[15];
  const float* dbih1 = (const float*)d_in[16];
  const float* dbhh1 = (const float*)d_in[17];
  const float* fcW = (const float*)d_in[18];
  const float* fcb = (const float*)d_in[19];
  const int* src = (const int*)d_in[20];
  const int* trg = (const int*)d_in[21];
  const int* tf_mask = (const int*)d_in[22];
  float* out = (float*)d_out;

  // Workspace carve (256-byte aligned slots)
  size_t off = 0;
  char* base = (char*)d_ws;
  auto carve = [&](size_t bytes) -> void* {
    void* p = base + off;
    off += (bytes + 255) & ~(size_t)255;
    return p;
  };
  f16* fcW16 = (f16*)carve((size_t)32000 * 512 * 2);
  f16* encEmb16 = (f16*)carve((size_t)32000 * 256 * 2);
  f16* decEmb16 = (f16*)carve((size_t)32000 * 256 * 2);
  f16* eW16ih0 = (f16*)carve((size_t)2048 * 256 * 2);
  f16* eW16hh0 = (f16*)carve((size_t)2048 * 512 * 2);
  f16* eW16ih1 = (f16*)carve((size_t)2048 * 512 * 2);
  f16* eW16hh1 = (f16*)carve((size_t)2048 * 512 * 2);
  f16* dW16ih0 = (f16*)carve((size_t)1536 * 256 * 2);
  f16* dW16hh0 = (f16*)carve((size_t)1536 * 512 * 2);
  f16* dW16ih1 = (f16*)carve((size_t)1536 * 512 * 2);
  f16* dW16hh1 = (f16*)carve((size_t)1536 * 512 * 2);
  float* h_enc = (float*)carve((size_t)2 * 32 * 512 * 4);
  float* c_enc = (float*)carve((size_t)2 * 32 * 512 * 4);
  float* h_dec = (float*)carve((size_t)2 * 32 * 512 * 4);
  f16* h16g0 = (f16*)carve((size_t)32 * 512 * 2);
  f16* h16g1 = (f16*)carve((size_t)32 * 512 * 2);
  int* tok = (int*)carve((size_t)32 * 4);

  auto cvt = [&](const float* s, f16* d, int n) {
    f32_to_f16_kernel<<<(n + 255) / 256, 256, 0, stream>>>(s, d, n);
  };
  cvt(fcW, fcW16, 32000 * 512);
  cvt(enc_emb, encEmb16, 32000 * 256);
  cvt(dec_emb, decEmb16, 32000 * 256);
  cvt(eWih0, eW16ih0, 2048 * 256);
  cvt(eWhh0, eW16hh0, 2048 * 512);
  cvt(eWih1, eW16ih1, 2048 * 512);
  cvt(eWhh1, eW16hh1, 2048 * 512);
  cvt(dWih0, dW16ih0, 1536 * 256);
  cvt(dWhh0, dW16hh0, 1536 * 512);
  cvt(dWih1, dW16ih1, 1536 * 512);
  cvt(dWhh1, dW16hh1, 1536 * 512);

  // outputs[0] stays zeros
  zero_kernel<<<(32 * 32000 + 255) / 256, 256, 0, stream>>>(out, 32 * 32000);

  const size_t rec_lds = (size_t)5 * 16384 * sizeof(f16);  // 160 KB
  encoder_kernel<<<1, 512, rec_lds, stream>>>(
      encEmb16, src, trg, eW16ih0, eW16hh0, eW16ih1, eW16hh1, ebih0, ebhh0,
      ebih1, ebhh1, h_enc, c_enc, h_dec, h16g0, h16g1, tok);

  for (int t = 1; t < 64; ++t) {
    dec_gru_kernel<<<1, 512, rec_lds, stream>>>(
        decEmb16, tok, dW16ih0, dW16hh0, dW16ih1, dW16hh1, dbih0, dbhh0, dbih1,
        dbhh1, h_dec, h16g0, h16g1);
    float* pred = out + (size_t)t * 32 * 32000;
    fc_kernel<<<500, 256, 0, stream>>>(h16g1, fcW16, fcb, pred);
    argmax_kernel<<<32, 256, 0, stream>>>(pred, trg, tf_mask, tok, t);
  }
}
